// CascadedGroupAttention_75453985457396
// MI455X (gfx1250) — matrix operand order repaired and checked by reading, not compiled, not run on hardware
//
#include <hip/hip_runtime.h>
#include <hip/hip_bf16.h>
#include <cstdint>
#include <cstddef>

typedef __attribute__((ext_vector_type(16))) __bf16 v16bf;
typedef __attribute__((ext_vector_type(8)))  __bf16 v8bf;
typedef __attribute__((ext_vector_type(8)))  float  v8f;

#define BLOCK 256
#define WPB   8   // waves per block (wave32)

union V16U { v16bf v; v8bf h[2]; };

__device__ __forceinline__ v8f wmma_bf16(v16bf a, v16bf b, v8f c) {
  // D(16x16,f32) = A(16x32,bf16) x B(32x16,bf16) + C
  return __builtin_amdgcn_wmma_f32_16x16x32_bf16(false, a, false, b, (short)0, c, false, false);
}

// A fragment: 16x32 tile of row-major [M][K]; ld = row stride (elems, mult of 8).
// ISA layout: lane L -> M=L%16; lanes 0-15: K=0..7 & 16..23; lanes 16-31: +8.
// -> two contiguous 16B runs per lane: [kh..kh+7] and [kh+16..kh+23].
__device__ __forceinline__ v16bf load_A(const __bf16* src, int ld, int lane) {
  int m  = lane & 15;
  int kh = (lane >> 4) * 8;
  const __bf16* p = src + m * ld + kh;
  V16U u;
  u.h[0] = *(const v8bf*)(p);
  u.h[1] = *(const v8bf*)(p + 16);
  return u.v;
}

// B fragment where memory holds B^T row-major: Bt[n][k], ld mult of 8.
// ISA layout: lane L -> N=L%16; lanes 0-15: K=0..7 & 16..23; lanes 16-31: +8.
// -> two contiguous 16B runs per lane: [kh..kh+7] and [kh+16..kh+23].
__device__ __forceinline__ v16bf load_Bt(const __bf16* src, int ld, int lane) {
  int n  = lane & 15;
  int kh = (lane >> 4) * 8;
  const __bf16* p = src + n * ld + kh;
  V16U u;
  u.h[0] = *(const v8bf*)(p);
  u.h[1] = *(const v8bf*)(p + 16);
  return u.v;
}

// ---------------------------------------------------------------------------
// Kernel 0: gather attention bias table, bf16-cast weights, fold BN.
// ---------------------------------------------------------------------------
__global__ void __launch_bounds__(256)
cga_prep_kernel(const float* __restrict__ qkv_w, const float* __restrict__ proj_w,
                const float* __restrict__ bn_gamma, const float* __restrict__ bn_beta,
                const float* __restrict__ bn_mean, const float* __restrict__ bn_var,
                const float* __restrict__ attn_biases, const int* __restrict__ bias_idxs,
                int n_off,
                float* __restrict__ abws, __bf16* __restrict__ qkvw,
                __bf16* __restrict__ projw,
                float* __restrict__ bnscale, float* __restrict__ bnshift)
{
  const int AB = 4 * 196 * 196;     // 153664
  const int QW = 4 * 96 * 64;       // 24576
  const int PW = 256 * 256;         // 65536
  int t = blockIdx.x * 256 + threadIdx.x;
  if (t < AB) {
    int h = t / 38416, p = t % 38416;
    abws[t] = attn_biases[h * n_off + bias_idxs[p]];
  } else if (t < AB + QW) {
    int i = t - AB;
    qkvw[i] = (__bf16)qkv_w[i];
  } else if (t < AB + QW + PW) {
    int i = t - (AB + QW);
    projw[i] = (__bf16)proj_w[i];
  } else if (t < AB + QW + PW + 256) {
    int o = t - (AB + QW + PW);
    float inv = bn_gamma[o] * rsqrtf(bn_var[o] + 1e-5f);
    bnscale[o] = inv;
    bnshift[o] = bn_beta[o] - bn_mean[o] * inv;
  }
}

// ---------------------------------------------------------------------------
// Kernel 1: cascaded 4-head attention, one workgroup per batch element.
// Produces cat = relu(concat(head outs)) as bf16 [B][208][256] (n-major) in ws.
// ---------------------------------------------------------------------------
__global__ void __launch_bounds__(256)
cga_attn_kernel(const float* __restrict__ x,
                const __bf16* __restrict__ qkvw,
                const float* __restrict__ qkv_b,
                const float* __restrict__ dw_w,
                const float* __restrict__ dw_b,
                const float* __restrict__ abws,
                __bf16* __restrict__ catp)
{
  __shared__ alignas(16) __bf16 s_featbT[208 * 64];  // qkv B^T: [n][c]
  __shared__ alignas(16) float  s_qraw[16 * 208];    // pre-conv q, [c][n]
  __shared__ alignas(16) __bf16 s_qT[208 * 32];      // q^T A-matrix [n][c], c pad 16->32
  __shared__ alignas(16) __bf16 s_kbT[208 * 32];     // k B^T [m][c], cols 16..31 zero
  __shared__ alignas(16) __bf16 s_vb[64 * 224];      // v A-matrix [d][m], m pad 196->224
  __shared__ alignas(16) float  s_S[16 * 208];       // score row-tile, f32
  __shared__ alignas(16) __bf16 s_P[16 * 224];       // softmax row-tile (P^T rows)
  __shared__ float  s_red[16 * 16];
  __shared__ float  s_dww[16 * 25];
  __shared__ float  s_dwb[16];
  __shared__ float  s_qb[96];

  const int tid  = threadIdx.x;
  const int lane = tid & 31;
  const int wave = tid >> 5;
  const int nloc = lane & 15;
  const int mh   = (lane >> 4) * 8;
  const int bix  = blockIdx.x;

  // ---- one-time init / zero padding ----
  for (int i = tid; i < 64 * 208; i += BLOCK) {       // coalesced read of x chunk 0
    int c = i / 208, n = i % 208;
    s_featbT[n * 64 + c] = (n < 196) ? (__bf16)x[((size_t)bix * 256 + c) * 196 + n]
                                     : (__bf16)0.0f;
  }
  for (int i = tid; i < 208 * 32; i += BLOCK) s_qT[i] = (__bf16)0.0f;
  for (int i = tid; i < 208 * 16; i += BLOCK) s_kbT[(i >> 4) * 32 + 16 + (i & 15)] = (__bf16)0.0f;
  for (int i = tid; i < 64 * 16; i += BLOCK)  s_vb[(i >> 4) * 224 + 208 + (i & 15)] = (__bf16)0.0f;
  for (int i = tid; i < 16 * 28; i += BLOCK)  s_P[(i / 28) * 224 + 196 + (i % 28)] = (__bf16)0.0f;
  __syncthreads();

  for (int head = 0; head < 4; ++head) {
    for (int i = tid; i < 400; i += BLOCK) s_dww[i] = dw_w[head * 400 + i];
    if (tid < 16) s_dwb[tid] = dw_b[head * 16 + tid];
    if (tid < 96) s_qb[tid] = qkv_b[head * 96 + tid];
    __syncthreads();

    // ---- QKV GEMM: f[96x196] = W[96x64] x feat[64x196] (+bias), routed ----
    const __bf16* wq = qkvw + head * 96 * 64;
    for (int j = wave; j < 78; j += WPB) {   // 6 M-tiles x 13 N-tiles
      int mt = j % 6, nt = j / 6;
      v8f acc = {0.f, 0.f, 0.f, 0.f, 0.f, 0.f, 0.f, 0.f};
#pragma unroll
      for (int kt = 0; kt < 2; ++kt)
        acc = wmma_bf16(load_A(wq + mt * 16 * 64 + kt * 32, 64, lane),
                        load_Bt(s_featbT + nt * 16 * 64 + kt * 32, 64, lane), acc);
      int n = nt * 16 + nloc;
      if (mt == 0) {                       // q rows 0..15 (f32, conv input)
#pragma unroll
        for (int v = 0; v < 8; ++v)
          s_qraw[(mh + v) * 208 + n] = acc[v] + s_qb[mh + v];
      } else if (mt == 1) {                // k rows 16..31 -> kb^T [m][c]
#pragma unroll
        for (int v = 0; v < 8; ++v)
          s_kbT[n * 32 + mh + v] = (__bf16)(acc[v] + s_qb[16 + mh + v]);
      } else {                             // v rows 32..95 -> vb [d][m]
        int dbase = (mt - 2) * 16;
#pragma unroll
        for (int v = 0; v < 8; ++v) {
          float val = acc[v] + s_qb[32 + dbase + mh + v];
          s_vb[(dbase + mh + v) * 224 + n] = (n < 196) ? (__bf16)val : (__bf16)0.0f;
        }
      }
    }
    __syncthreads();

    // ---- 5x5 depthwise conv on q (SAME, 14x14), write transposed bf16 ----
    for (int i = tid; i < 3136; i += BLOCK) {
      int c = i & 15, n = i >> 4;
      int py = n / 14, px = n % 14;
      float s = s_dwb[c];
#pragma unroll
      for (int dy = -2; dy <= 2; ++dy) {
        int yy = py + dy;
        if (yy < 0 || yy >= 14) continue;
#pragma unroll
        for (int dx = -2; dx <= 2; ++dx) {
          int xx = px + dx;
          if (xx < 0 || xx >= 14) continue;
          s += s_qraw[c * 208 + yy * 14 + xx] * s_dww[c * 25 + (dy + 2) * 5 + (dx + 2)];
        }
      }
      s_qT[n * 32 + c] = (__bf16)s;
    }
    __syncthreads();

    // ---- attention, 16 query-rows (one row-tile) at a time ----
    const float* abh = abws + head * 38416;
    for (int rt = 0; rt < 13; ++rt) {
      // scores: S[16 x 196] = qT(rows rt) x k   (K=32, upper 16 zero-padded)
      v16bf aq = load_A(s_qT + rt * 16 * 32, 32, lane);
      for (int ct = wave; ct < 13; ct += WPB) {
        v8f acc = {0.f, 0.f, 0.f, 0.f, 0.f, 0.f, 0.f, 0.f};
        acc = wmma_bf16(aq, load_Bt(s_kbT + ct * 16 * 32, 32, lane), acc);
#pragma unroll
        for (int v = 0; v < 8; ++v)
          s_S[(mh + v) * 208 + ct * 16 + nloc] = acc[v];
      }
      __syncthreads();

      // softmax over 196 cols (scale = KEY_DIM^-0.5 = 0.25, + positional bias)
      {
        int r = tid >> 4, c0 = tid & 15;
        int nr = rt * 16 + r;
        float lmax = -3.0e38f;
        for (int m = c0; m < 196; m += 16) {
          float val = s_S[r * 208 + m] * 0.25f + ((nr < 196) ? abh[nr * 196 + m] : 0.f);
          s_S[r * 208 + m] = val;
          lmax = fmaxf(lmax, val);
        }
        s_red[r * 16 + c0] = lmax;
        __syncthreads();
        float rmax = -3.0e38f;
#pragma unroll
        for (int t2 = 0; t2 < 16; ++t2) rmax = fmaxf(rmax, s_red[r * 16 + t2]);
        __syncthreads();
        float lsum = 0.f;
        for (int m = c0; m < 196; m += 16) {
          float e = __expf(s_S[r * 208 + m] - rmax);
          s_S[r * 208 + m] = e;
          lsum += e;
        }
        s_red[r * 16 + c0] = lsum;
        __syncthreads();
        float rsum = 0.f;
#pragma unroll
        for (int t2 = 0; t2 < 16; ++t2) rsum += s_red[r * 16 + t2];
        float inv = 1.0f / rsum;
        for (int m = c0; m < 196; m += 16)
          s_P[r * 224 + m] = (__bf16)(s_S[r * 208 + m] * inv);
      }
      __syncthreads();

      // AV: out[:, n in rt] (64x16) = v(64x224) x P^T(224x16); K=224 padded
      if (wave < 4) {
        int mt = wave;
        v8f acc = {0.f, 0.f, 0.f, 0.f, 0.f, 0.f, 0.f, 0.f};
#pragma unroll
        for (int kt = 0; kt < 7; ++kt)
          acc = wmma_bf16(load_A(s_vb + mt * 16 * 224 + kt * 32, 224, lane),
                          load_Bt(s_P + kt * 32, 224, lane), acc);
        int n = rt * 16 + nloc;
        if (n < 196) {
#pragma unroll
          for (int v = 0; v < 8; ++v) {
            int d = mt * 16 + mh + v;
            float val = acc[v];
            // cat stored n-major (B^T for proj): [b][n][c]
            catp[((size_t)bix * 208 + n) * 256 + head * 64 + d] = (__bf16)fmaxf(val, 0.f);
            if (head < 3)
              s_featbT[n * 64 + d] =
                  (__bf16)(val + x[((size_t)bix * 256 + (head + 1) * 64 + d) * 196 + n]);
          }
        }
      }
      __syncthreads();
    }
  }
}

// ---------------------------------------------------------------------------
// Kernel 2: out[b][o][n] = bnscale[o] * (proj_w[o,:] @ cat[b][n][:]) + bnshift[o]
// ---------------------------------------------------------------------------
__global__ void __launch_bounds__(256)
cga_proj_kernel(const __bf16* __restrict__ projw, const __bf16* __restrict__ catp,
                const float* __restrict__ bnscale, const float* __restrict__ bnshift,
                float* __restrict__ out)
{
  const int bix  = blockIdx.x;
  const int half = blockIdx.y;
  const int lane = threadIdx.x & 31;
  const int wave = threadIdx.x >> 5;
  const int nloc = lane & 15;
  const int mh   = (lane >> 4) * 8;
  const __bf16* catb = catp + (size_t)bix * 208 * 256;   // [n][c]

  for (int j = wave; j < 104; j += WPB) {   // 8 M-tiles x 13 N-tiles per block
    int mt = half * 8 + (j & 7);
    int nt = j >> 3;
    v8f acc = {0.f, 0.f, 0.f, 0.f, 0.f, 0.f, 0.f, 0.f};
#pragma unroll
    for (int kt = 0; kt < 8; ++kt)
      acc = wmma_bf16(load_A(projw + mt * 16 * 256 + kt * 32, 256, lane),
                      load_Bt(catb + nt * 16 * 256 + kt * 32, 256, lane), acc);
    int n = nt * 16 + nloc;
    if (n < 196) {
#pragma unroll
      for (int v = 0; v < 8; ++v) {
        int o = mt * 16 + mh + v;
        out[((size_t)bix * 256 + o) * 196 + n] = acc[v] * bnscale[o] + bnshift[o];
      }
    }
  }
}

// ---------------------------------------------------------------------------
extern "C" void kernel_launch(void* const* d_in, const int* in_sizes, int n_in,
                              void* d_out, int out_size, void* d_ws, size_t ws_size,
                              hipStream_t stream) {
  (void)n_in; (void)out_size; (void)ws_size;
  const float* x      = (const float*)d_in[0];
  const float* qkv_w  = (const float*)d_in[1];
  const float* qkv_b  = (const float*)d_in[2];
  const float* dw_w   = (const float*)d_in[3];
  const float* dw_b   = (const float*)d_in[4];
  const float* proj_w = (const float*)d_in[5];
  const float* bn_g   = (const float*)d_in[6];
  const float* bn_b   = (const float*)d_in[7];
  const float* bn_m   = (const float*)d_in[8];
  const float* bn_v   = (const float*)d_in[9];
  const float* att_b  = (const float*)d_in[10];
  const int*   bidx   = (const int*)d_in[11];

  const int Bsz   = in_sizes[0] / (256 * 196);
  const int n_off = in_sizes[10] / 4;

  // workspace layout (bytes); all segments 16B-aligned
  char* ws = (char*)d_ws;
  float*  abws    = (float*)(ws + 0);        // 4*196*196*4 = 614656
  __bf16* qkvw    = (__bf16*)(ws + 614656);  // 4*96*64*2   =  49152
  __bf16* projw   = (__bf16*)(ws + 663808);  // 256*256*2   = 131072
  float*  bnscale = (float*)(ws + 794880);   // 256*4
  float*  bnshift = (float*)(ws + 795904);   // 256*4
  __bf16* catp    = (__bf16*)(ws + 796928);  // B*208*256*2

  const int prep_total = 4 * 196 * 196 + 4 * 96 * 64 + 256 * 256 + 256;
  hipLaunchKernelGGL(cga_prep_kernel, dim3((prep_total + 255) / 256), dim3(256), 0, stream,
                     qkv_w, proj_w, bn_g, bn_b, bn_m, bn_v, att_b, bidx, n_off,
                     abws, qkvw, projw, bnscale, bnshift);
  hipLaunchKernelGGL(cga_attn_kernel, dim3(Bsz), dim3(256), 0, stream,
                     x, qkvw, qkv_b, dw_w, dw_b, abws, catp);
  hipLaunchKernelGGL(cga_proj_kernel, dim3(Bsz, 2), dim3(256), 0, stream,
                     projw, catp, bnscale, bnshift, (float*)d_out);
}
